// MambaBlock_40037685133981
// MI455X (gfx1250) — matrix operand order repaired and checked by reading, not compiled, not run on hardware
//
#include <hip/hip_runtime.h>
#include <hip/hip_bf16.h>
#include <math.h>

// ---------------------------------------------------------------------------
// MI455X (gfx1250) fused Mamba-window block.
//  k1: gather + LN + loan1 + in_proj (WMMA f16, 128->256) -> h
//  k2: per-sequence mamba (conv3+silu, xproj, dt, softplus, scalar scan) -> yf/yb
//  k3: combine + out_proj (WMMA) + residual gather -> x1
//  k4: LN + loan2 + fc1 (WMMA) + exact GeLU -> g (f16)
//  k5: fc2 (WMMA) + bias + residual -> x2
//  k6: gather c1 -> out
// ---------------------------------------------------------------------------

typedef __attribute__((ext_vector_type(16))) _Float16 v16h;
typedef __attribute__((ext_vector_type(8)))  float    v8f;

namespace {
constexpr int Bq  = 2;
constexpr int Tq  = 6;
constexpr int Pq  = 10000;
constexpr int DIMq  = 128;
constexpr int DIMSq = 16;
constexpr int G0q = 3;
constexpr int G1q = 10;
constexpr int DINq = 128;              // D_INNER
constexpr int Lq   = G0q * G1q;        // 30
constexpr int TBq  = Tq / G0q;         // 2
constexpr int PBq  = Pq / G1q;         // 1000
constexpr int NSEQ = Bq * TBq * PBq;   // 4000 per direction
constexpr int NROWS = Bq * Tq * Pq;    // 120000
constexpr int RPB = 64;                // rows per GEMM block
constexpr int ASTR = 136;              // LDS A stride (f16), 16B aligned, padded
}

__device__ __forceinline__ void row_decode(int row, int& b, int& t, int& p) {
  b = row / (Tq * Pq);
  int r = row - b * (Tq * Pq);
  t = r / Pq;
  p = r - t * Pq;
}

// A fragment (16x32 f16) from LDS, ISA 7.12.2 layout:
//   M = lane%16 ; half i<8 -> K = kk*32 + (lane/16)*8 + i
//                 half i>=8 -> K = kk*32 + 16 + (lane/16)*8 + (i-8)
__device__ __forceinline__ v16h load_a_frag(const _Float16* As, int m, int kk, int lane) {
  const _Float16* base = As + m * ASTR + kk * 32 + ((lane >> 4) << 3);
  v16h a;
#pragma unroll
  for (int i = 0; i < 8; ++i) {
    a[i]     = base[i];
    a[i + 8] = base[16 + i];
  }
  return a;
}

// B fragment (32x16 f16) from f32 weight matrix W[N][K] (out-features major):
//   N = n ; half i<8 -> K = kk*32 + (lane/16)*8 + i ; i>=8 -> K = kk*32 + 16 + (lane/16)*8 + (i-8)
__device__ __forceinline__ v16h load_b_frag(const float* W, int K, int n, int kk, int lane) {
  const float* base = W + (size_t)n * K + kk * 32 + ((lane >> 4) << 3);
  v16h b;
#pragma unroll
  for (int i = 0; i < 8; ++i) { b[i] = (_Float16)base[i]; b[i + 8] = (_Float16)base[16 + i]; }
  return b;
}

__device__ __forceinline__ float silu_f(float x) { return x / (1.f + expf(-x)); }

// ---------------------------------------------------------------------------
// k1: gather(c0) + LN + loan1 + in_proj (128 -> 256)
// ---------------------------------------------------------------------------
__global__ __launch_bounds__(256) void k_inproj(const float* __restrict__ x,
                                                const float* __restrict__ xstat,
                                                const int*   __restrict__ curves,
                                                const float* __restrict__ l1w,
                                                const float* __restrict__ l1b,
                                                const float* __restrict__ Win,
                                                float* __restrict__ hbuf) {
  __shared__ float    xr[RPB][DIMq + 4];
  __shared__ float    ssr[RPB][DIMSq];
  __shared__ float    mean_s[RPB], inv_s[RPB];
  __shared__ _Float16 As[RPB][ASTR];

  const int tid  = threadIdx.x;
  const int row0 = blockIdx.x * RPB;

  for (int idx = tid; idx < RPB * DIMq; idx += 256) {
    int r = idx >> 7, c = idx & 127;
    int b, t, p; row_decode(row0 + r, b, t, p);
    int ps = curves[b * 2 * Pq + p];
    xr[r][c] = x[((size_t)((b * Tq + t) * Pq + ps)) * DIMq + c];
  }
  for (int idx = tid; idx < RPB * DIMSq; idx += 256) {
    int r = idx >> 4, c = idx & 15;
    int b, t, p; row_decode(row0 + r, b, t, p);
    int ps = curves[b * 2 * Pq + p];
    ssr[r][c] = xstat[((size_t)((b * Tq + t) * Pq + ps)) * DIMSq + c];
  }
  __syncthreads();

  if (tid < RPB) {
    float s = 0.f, s2 = 0.f;
    for (int c = 0; c < DIMq; ++c) { float v = xr[tid][c]; s += v; s2 += v * v; }
    float m = s * (1.f / DIMq);
    float var = s2 * (1.f / DIMq) - m * m;
    mean_s[tid] = m;
    inv_s[tid]  = rsqrtf(var + 1e-5f);
  }
  __syncthreads();

  for (int idx = tid; idx < RPB * DIMq; idx += 256) {
    int r = idx >> 7, c = idx & 127;
    float acc = l1b[c];
#pragma unroll
    for (int k = 0; k < DIMSq; ++k) acc += ssr[r][k] * l1w[c * DIMSq + k];
    float hv = (xr[r][c] - mean_s[r]) * inv_s[r] + acc;
    As[r][c] = (_Float16)hv;
  }
  __syncthreads();

  const int lane = tid & 31, wave = tid >> 5;
  v16h bf[2][4];
#pragma unroll
  for (int j = 0; j < 2; ++j) {
    int nt = wave + j * 8;
#pragma unroll
    for (int kk = 0; kk < 4; ++kk)
      bf[j][kk] = load_b_frag(Win, DIMq, nt * 16 + (lane & 15), kk, lane);
  }

#pragma unroll
  for (int mt = 0; mt < 4; ++mt) {
    v16h a[4];
#pragma unroll
    for (int kk = 0; kk < 4; ++kk)
      a[kk] = load_a_frag(&As[0][0], mt * 16 + (lane & 15), kk, lane);
#pragma unroll
    for (int j = 0; j < 2; ++j) {
      v8f c = {};
#pragma unroll
      for (int kk = 0; kk < 4; ++kk)
        c = __builtin_amdgcn_wmma_f32_16x16x32_f16(false, a[kk], false, bf[j][kk],
                                                   (short)0, c, false, false);
      int n = (wave + j * 8) * 16 + (lane & 15);
#pragma unroll
      for (int rr = 0; rr < 8; ++rr) {
        int m = mt * 16 + rr + ((lane >> 4) << 3);
        hbuf[(size_t)(row0 + m) * 256 + n] = c[rr];
      }
    }
  }
}

// ---------------------------------------------------------------------------
// k2: mamba scan per sequence. block = 128 threads (thread == channel).
// ---------------------------------------------------------------------------
__global__ __launch_bounds__(128) void k_mamba(const float* __restrict__ hbuf,
                                               const float* __restrict__ conv_w,
                                               const float* __restrict__ conv_b,
                                               const float* __restrict__ xproj_w,
                                               const float* __restrict__ dt_w,
                                               const float* __restrict__ dt_b,
                                               const float* __restrict__ A_log,
                                               const float* __restrict__ Dp,
                                               float* __restrict__ yf,
                                               float* __restrict__ yb) {
  __shared__ float u_s[Lq][DINq];
  __shared__ float z_s[Lq][DINq];
  __shared__ float dbl_s[Lq][10];

  const int ch  = threadIdx.x;
  const int s   = blockIdx.x;
  const int dir = s / NSEQ;
  const int r   = s - dir * NSEQ;
  const int b   = r / (TBq * PBq);
  const int tb  = (r / PBq) % TBq;
  const int pb  = r % PBq;

  const float w0 = conv_w[(dir * DINq + ch) * 3 + 0];
  const float w1 = conv_w[(dir * DINq + ch) * 3 + 1];
  const float w2 = conv_w[(dir * DINq + ch) * 3 + 2];
  const float cb = conv_b[dir * DINq + ch];

  float xm2 = 0.f, xm1 = 0.f;
  for (int l = 0; l < Lq; ++l) {
    int g0 = l / G1q, g1 = l % G1q;
    int t  = tb * G0q + g0;
    int pi = pb * G1q + g1;
    if (dir) pi = Pq - 1 - pi;
    const float* hp = hbuf + ((size_t)((b * Tq + t) * Pq + pi)) * 256;
    float xv = hp[ch];
    z_s[l][ch] = hp[DINq + ch];
    float xc = w0 * xm2 + w1 * xm1 + w2 * xv + cb;
    u_s[l][ch] = silu_f(xc);
    xm2 = xm1; xm1 = xv;
  }
  __syncthreads();

  for (int pr = ch; pr < Lq * 10; pr += DINq) {
    int t = pr / 10, rr = pr - t * 10;
    const float* wp = xproj_w + (size_t)(dir * 10 + rr) * DINq;
    float acc = 0.f;
    for (int c = 0; c < DINq; ++c) acc += u_s[t][c] * wp[c];
    dbl_s[t][rr] = acc;
  }
  __syncthreads();

  float dtw_r[8];
#pragma unroll
  for (int k = 0; k < 8; ++k) dtw_r[k] = dt_w[(dir * DINq + ch) * 8 + k];
  const float dtbv = dt_b[dir * DINq + ch];
  const float Av   = -expf(A_log[dir * DINq + ch]);   // D_STATE == 1
  const float Dv   = Dp[dir * DINq + ch];

  float* dst = dir ? yb : yf;
  float hstate = 0.f;
  for (int l = 0; l < Lq; ++l) {
    float dv = dtbv;
#pragma unroll
    for (int k = 0; k < 8; ++k) dv += dbl_s[l][k] * dtw_r[k];
    float delta = (dv > 20.f) ? dv : log1pf(expf(dv));       // softplus
    float Bm = dbl_s[l][8], Cm = dbl_s[l][9];
    float uv = u_s[l][ch];
    float dA = expf(delta * Av);
    hstate = dA * hstate + delta * uv * Bm;
    float y = hstate * Cm + uv * Dv;
    y *= silu_f(z_s[l][ch]);

    // exact replication of the reference's _wrev index scramble (+ final flip
    // for the backward direction):
    int g0 = l / G1q, g1 = l % G1q;
    int F  = tb * 30000 + pb * 30 + g0 * 10 + g1;   // flat within batch
    int i1 = F / 60;
    int i2 = (F / 30) & 1;
    int i3 = (F / 3) % 10;
    int i4 = F % 3;
    int J  = ((i3 * PBq + i1) * G0q + i4) * 2 + i2;
    int to = J / Pq;
    int po = J - to * Pq;
    int pw = dir ? (Pq - 1 - po) : po;
    dst[((size_t)((b * Tq + to) * Pq + pw)) * DINq + ch] = y;
  }
}

// ---------------------------------------------------------------------------
// k3: y = 0.5*(yf+yb) ; x1 = gather_x + y @ out_proj^T   (WMMA 128->128)
// ---------------------------------------------------------------------------
__global__ __launch_bounds__(256) void k_outproj(const float* __restrict__ yf,
                                                 const float* __restrict__ yb,
                                                 const float* __restrict__ Wout,
                                                 const float* __restrict__ x,
                                                 const int*   __restrict__ curves,
                                                 float* __restrict__ x1) {
  __shared__ _Float16 As[RPB][ASTR];
  const int tid = threadIdx.x;
  const int row0 = blockIdx.x * RPB;

  for (int idx = tid; idx < RPB * DIMq; idx += 256) {
    int r = idx >> 7, c = idx & 127;
    size_t off = (size_t)(row0 + r) * DIMq + c;
    As[r][c] = (_Float16)(0.5f * (yf[off] + yb[off]));
  }
  __syncthreads();

  const int lane = tid & 31, wave = tid >> 5;
  v16h bf[4];
#pragma unroll
  for (int kk = 0; kk < 4; ++kk)
    bf[kk] = load_b_frag(Wout, DIMq, wave * 16 + (lane & 15), kk, lane);

#pragma unroll
  for (int mt = 0; mt < 4; ++mt) {
    v16h a[4];
#pragma unroll
    for (int kk = 0; kk < 4; ++kk)
      a[kk] = load_a_frag(&As[0][0], mt * 16 + (lane & 15), kk, lane);
    v8f c = {};
#pragma unroll
    for (int kk = 0; kk < 4; ++kk)
      c = __builtin_amdgcn_wmma_f32_16x16x32_f16(false, a[kk], false, bf[kk],
                                                 (short)0, c, false, false);
    int n = wave * 16 + (lane & 15);
#pragma unroll
    for (int rr = 0; rr < 8; ++rr) {
      int m = mt * 16 + rr + ((lane >> 4) << 3);
      int row = row0 + m;
      int b, t, p; row_decode(row, b, t, p);
      int ps = curves[b * 2 * Pq + p];
      float xs = x[((size_t)((b * Tq + t) * Pq + ps)) * DIMq + n];
      x1[(size_t)row * DIMq + n] = c[rr] + xs;
    }
  }
}

// ---------------------------------------------------------------------------
// k4: LN(x1) + loan2 ; fc1 (WMMA) + bias ; exact GeLU -> g (f16)
// ---------------------------------------------------------------------------
__global__ __launch_bounds__(256) void k_fc1(const float* __restrict__ x1,
                                             const float* __restrict__ xstat,
                                             const int*   __restrict__ curves,
                                             const float* __restrict__ l2w,
                                             const float* __restrict__ l2b,
                                             const float* __restrict__ W1,
                                             const float* __restrict__ b1,
                                             _Float16* __restrict__ g16) {
  __shared__ float    xr[RPB][DIMq + 4];
  __shared__ float    ssr[RPB][DIMSq];
  __shared__ float    mean_s[RPB], inv_s[RPB];
  __shared__ _Float16 As[RPB][ASTR];

  const int tid = threadIdx.x;
  const int row0 = blockIdx.x * RPB;

  for (int idx = tid; idx < RPB * DIMq; idx += 256) {
    int r = idx >> 7, c = idx & 127;
    xr[r][c] = x1[(size_t)(row0 + r) * DIMq + c];
  }
  for (int idx = tid; idx < RPB * DIMSq; idx += 256) {
    int r = idx >> 4, c = idx & 15;
    int b, t, p; row_decode(row0 + r, b, t, p);
    int ps = curves[b * 2 * Pq + p];
    ssr[r][c] = xstat[((size_t)((b * Tq + t) * Pq + ps)) * DIMSq + c];
  }
  __syncthreads();

  if (tid < RPB) {
    float s = 0.f, s2 = 0.f;
    for (int c = 0; c < DIMq; ++c) { float v = xr[tid][c]; s += v; s2 += v * v; }
    float m = s * (1.f / DIMq);
    float var = s2 * (1.f / DIMq) - m * m;
    mean_s[tid] = m;
    inv_s[tid]  = rsqrtf(var + 1e-5f);
  }
  __syncthreads();

  for (int idx = tid; idx < RPB * DIMq; idx += 256) {
    int r = idx >> 7, c = idx & 127;
    float acc = l2b[c];
#pragma unroll
    for (int k = 0; k < DIMSq; ++k) acc += ssr[r][k] * l2w[c * DIMSq + k];
    As[r][c] = (_Float16)((xr[r][c] - mean_s[r]) * inv_s[r] + acc);
  }
  __syncthreads();

  const int lane = tid & 31, wave = tid >> 5;
  v16h bf[4];
#pragma unroll
  for (int kk = 0; kk < 4; ++kk)
    bf[kk] = load_b_frag(W1, DIMq, wave * 16 + (lane & 15), kk, lane);

#pragma unroll
  for (int mt = 0; mt < 4; ++mt) {
    v16h a[4];
#pragma unroll
    for (int kk = 0; kk < 4; ++kk)
      a[kk] = load_a_frag(&As[0][0], mt * 16 + (lane & 15), kk, lane);
    v8f c = {};
#pragma unroll
    for (int kk = 0; kk < 4; ++kk)
      c = __builtin_amdgcn_wmma_f32_16x16x32_f16(false, a[kk], false, bf[kk],
                                                 (short)0, c, false, false);
    int n = wave * 16 + (lane & 15);
#pragma unroll
    for (int rr = 0; rr < 8; ++rr) {
      int m = mt * 16 + rr + ((lane >> 4) << 3);
      float v = c[rr] + b1[n];
      float g = 0.5f * v * (1.f + erff(v * 0.70710678118654752f));  // exact GeLU
      g16[(size_t)(row0 + m) * DIMq + n] = (_Float16)g;
    }
  }
}

// ---------------------------------------------------------------------------
// k5: x2 = x1 + g @ fc2^T + fc2_b   (WMMA 128->128)
// ---------------------------------------------------------------------------
__global__ __launch_bounds__(256) void k_fc2(const _Float16* __restrict__ g16,
                                             const float* __restrict__ W2,
                                             const float* __restrict__ b2,
                                             const float* __restrict__ x1,
                                             float* __restrict__ x2) {
  __shared__ _Float16 As[RPB][ASTR];
  const int tid = threadIdx.x;
  const int row0 = blockIdx.x * RPB;

  for (int idx = tid; idx < RPB * DIMq; idx += 256) {
    int r = idx >> 7, c = idx & 127;
    As[r][c] = g16[(size_t)(row0 + r) * DIMq + c];
  }
  __syncthreads();

  const int lane = tid & 31, wave = tid >> 5;
  v16h bf[4];
#pragma unroll
  for (int kk = 0; kk < 4; ++kk)
    bf[kk] = load_b_frag(W2, DIMq, wave * 16 + (lane & 15), kk, lane);

#pragma unroll
  for (int mt = 0; mt < 4; ++mt) {
    v16h a[4];
#pragma unroll
    for (int kk = 0; kk < 4; ++kk)
      a[kk] = load_a_frag(&As[0][0], mt * 16 + (lane & 15), kk, lane);
    v8f c = {};
#pragma unroll
    for (int kk = 0; kk < 4; ++kk)
      c = __builtin_amdgcn_wmma_f32_16x16x32_f16(false, a[kk], false, bf[kk],
                                                 (short)0, c, false, false);
    int n = wave * 16 + (lane & 15);
#pragma unroll
    for (int rr = 0; rr < 8; ++rr) {
      int m = mt * 16 + rr + ((lane >> 4) << 3);
      size_t off = (size_t)(row0 + m) * DIMq + n;
      x2[off] = c[rr] + b2[n] + x1[off];
    }
  }
}

// ---------------------------------------------------------------------------
// k6: out[b,t,p,:] = x2[b,t,c1[b,p],:]
// ---------------------------------------------------------------------------
__global__ __launch_bounds__(256) void k_gather(const float* __restrict__ x2,
                                                const int*   __restrict__ curves,
                                                float* __restrict__ out) {
  int idx = blockIdx.x * 256 + threadIdx.x;
  if (idx >= NROWS * DIMq) return;
  int c = idx & 127;
  int row = idx >> 7;
  int b, t, p; row_decode(row, b, t, p);
  int ps = curves[b * 2 * Pq + Pq + p];   // c1
  out[idx] = x2[((size_t)((b * Tq + t) * Pq + ps)) * DIMq + c];
}

// ---------------------------------------------------------------------------
extern "C" void kernel_launch(void* const* d_in, const int* in_sizes, int n_in,
                              void* d_out, int out_size, void* d_ws, size_t ws_size,
                              hipStream_t stream) {
  const float* x        = (const float*)d_in[0];
  const float* xstat    = (const float*)d_in[1];
  const int*   curves   = (const int*)  d_in[2];
  const float* loan1_w  = (const float*)d_in[3];
  const float* loan1_b  = (const float*)d_in[4];
  const float* loan2_w  = (const float*)d_in[5];
  const float* loan2_b  = (const float*)d_in[6];
  const float* in_proj  = (const float*)d_in[7];
  const float* out_proj = (const float*)d_in[8];
  const float* conv_w   = (const float*)d_in[9];
  const float* conv_b   = (const float*)d_in[10];
  const float* xproj_w  = (const float*)d_in[11];
  const float* dt_w     = (const float*)d_in[12];
  const float* dt_b     = (const float*)d_in[13];
  const float* A_log    = (const float*)d_in[14];
  const float* Dvec     = (const float*)d_in[15];
  const float* fc1_w    = (const float*)d_in[16];
  const float* fc1_b    = (const float*)d_in[17];
  const float* fc2_w    = (const float*)d_in[18];
  const float* fc2_b    = (const float*)d_in[19];
  float* out = (float*)d_out;

  // Workspace layout (floats). Peak = 245.76 MB.
  //   region0: h  (NROWS*256)  -> reused as x1 after k2
  //   region1: yf (NROWS*128)  -> reused as g16 (f16, half the bytes) after k3
  //   region2: yb (NROWS*128)  -> reused as x2 after k4
  float* hbuf = (float*)d_ws;
  float* yfb  = hbuf + (size_t)NROWS * 256;
  float* ybb  = yfb + (size_t)NROWS * DIMq;
  float* x1b  = hbuf;                 // overwrites h (dead after k2)
  _Float16* g16 = (_Float16*)yfb;     // overwrites yf (dead after k3)
  float* x2b  = ybb;                  // overwrites yb (dead after k3)

  const int gemm_blocks = NROWS / RPB;   // 1875 (exact)

  k_inproj<<<gemm_blocks, 256, 0, stream>>>(x, xstat, curves, loan1_w, loan1_b,
                                            in_proj, hbuf);
  k_mamba<<<2 * NSEQ, 128, 0, stream>>>(hbuf, conv_w, conv_b, xproj_w, dt_w, dt_b,
                                        A_log, Dvec, yfb, ybb);
  k_outproj<<<gemm_blocks, 256, 0, stream>>>(yfb, ybb, out_proj, x, curves, x1b);
  k_fc1<<<gemm_blocks, 256, 0, stream>>>(x1b, xstat, curves, loan2_w, loan2_b,
                                         fc1_w, fc1_b, g16);
  k_fc2<<<gemm_blocks, 256, 0, stream>>>(g16, fc2_w, fc2_b, x1b, x2b);
  k_gather<<<(NROWS * DIMq + 255) / 256, 256, 0, stream>>>(x2b, curves, out);
}
